// NTXentLoss_41266045780374
// MI455X (gfx1250) — matrix operand order repaired and checked by reading, not compiled, not run on hardware
//
#include <hip/hip_runtime.h>

typedef _Float16 v16h __attribute__((ext_vector_type(16)));
typedef _Float16 v8h  __attribute__((ext_vector_type(8)));
typedef _Float16 v4h  __attribute__((ext_vector_type(4)));
typedef float    v8f  __attribute__((ext_vector_type(8)));
typedef unsigned int u32x4 __attribute__((ext_vector_type(4)));
typedef int      i32x8 __attribute__((ext_vector_type(8)));
typedef int      i32x4 __attribute__((ext_vector_type(4)));

#define NROWS  8192   // 2N
#define HALF_N 4096
#define DDIM   128

// ---------------------------------------------------------------------------
// Tensor Data Mover: load one 16-row x 256-byte tile of r into LDS.
// D# built per CDNA5 ISA 8.3/8.4: data_size=4B, tensor_dim0=64 dwords,
// tensor_dim1=8192 rows, tile=64x16, row stride=64 dwords.
// This toolchain exposes the 6-arg builtin:
//   (v4u group0, v8i group1, v4i group2, v4i group3, v8i extra, i32 cpol)
// Groups 2/3 + extra are zero (2D tensor, no higher dims).
// ---------------------------------------------------------------------------
__device__ __forceinline__ void tdm_load_tile(unsigned long long gaddr,
                                              unsigned lds_addr) {
  u32x4 g0;
  g0[0] = 1u;                                   // count=1, user descriptor
  g0[1] = lds_addr;                             // LDS byte address
  g0[2] = (unsigned)(gaddr & 0xffffffffull);    // global_addr[31:0]
  g0[3] = (unsigned)((gaddr >> 32) & 0x01ffffffull) | (2u << 30); // addr[56:32], type=2
  i32x8 g1;
  g1[0] = (int)(2u << 16);          // workgroup_mask=0, data_size=2 (4 bytes)
  g1[1] = (int)(64u << 16);         // tensor_dim0 = 64 (dwords per row)
  g1[2] = (int)(8192u << 16);       // tensor_dim0 hi=0, tensor_dim1 = 8192
  g1[3] = (int)(64u << 16);         // tensor_dim1 hi=0, tile_dim0 = 64
  g1[4] = 16;                       // tile_dim1 = 16 rows, tile_dim2 = 0
  g1[5] = 64;                       // tensor_dim0_stride = 64 dwords
  g1[6] = (int)(64u << 16);         // stride hi=0, tensor_dim1_stride lo = 64
  g1[7] = 0;                        // tensor_dim1_stride hi
  const i32x4 gz4 = {0, 0, 0, 0};
  const i32x8 gz8 = {0, 0, 0, 0, 0, 0, 0, 0};
  __builtin_amdgcn_tensor_load_to_lds(g0, g1, gz4, gz4, gz8, 0);
}

// ---------------------------------------------------------------------------
// Kernel A: L2-normalize each of the 8192 rows, emit f16 r matrix (2 MiB).
// One wave per row, 4 floats per lane.
// ---------------------------------------------------------------------------
__global__ __launch_bounds__(256) void ntx_normalize(const float* __restrict__ zi,
                                                     const float* __restrict__ zj,
                                                     _Float16* __restrict__ r) {
  const int row  = blockIdx.x * 8 + (threadIdx.x >> 5);
  const int lane = threadIdx.x & 31;
  const float* src = (row < HALF_N) ? (zi + (size_t)row * DDIM)
                                    : (zj + (size_t)(row - HALF_N) * DDIM);
  const float4 x = ((const float4*)src)[lane];
  float ss = x.x * x.x + x.y * x.y + x.z * x.z + x.w * x.w;
#pragma unroll
  for (int m = 1; m <= 16; m <<= 1) ss += __shfl_xor(ss, m, 32);
  const float inv = 1.0f / fmaxf(sqrtf(ss), 1e-12f);
  v4h o;
  o[0] = (_Float16)(x.x * inv);
  o[1] = (_Float16)(x.y * inv);
  o[2] = (_Float16)(x.z * inv);
  o[3] = (_Float16)(x.w * inv);
  *(v4h*)(r + (size_t)row * DDIM + lane * 4) = o;
}

// ---------------------------------------------------------------------------
// Kernel B: pos_k = <r_k, r_{k^N}>, diag_k = <r_k, r_k> (f32 accum over f16).
// One wave per row.
// ---------------------------------------------------------------------------
__global__ __launch_bounds__(256) void ntx_posdiag(const _Float16* __restrict__ r,
                                                   float* __restrict__ pos,
                                                   float* __restrict__ dk) {
  const int k    = blockIdx.x * 8 + (threadIdx.x >> 5);
  const int lane = threadIdx.x & 31;
  const int p    = k ^ HALF_N;
  const v4h a = *(const v4h*)(r + (size_t)k * DDIM + lane * 4);
  const v4h b = *(const v4h*)(r + (size_t)p * DDIM + lane * 4);
  float sp = 0.f, sd = 0.f;
#pragma unroll
  for (int e = 0; e < 4; ++e) {
    const float av = (float)a[e], bv = (float)b[e];
    sp += av * bv;
    sd += av * av;
  }
#pragma unroll
  for (int m = 1; m <= 16; m <<= 1) {
    sp += __shfl_xor(sp, m, 32);
    sd += __shfl_xor(sd, m, 32);
  }
  if (lane == 0) { pos[k] = sp; dk[k] = sd; }
}

// ---------------------------------------------------------------------------
// Kernel C: fused sim GEMM + row exp-sums.
// Block = 8 waves, each wave owns a 16-row strip; block covers 2048 columns
// (one of 4 splits). B tiles stream through double-buffered LDS via the TDM.
// ---------------------------------------------------------------------------
__global__ __launch_bounds__(256) void ntx_simsum(const _Float16* __restrict__ r,
                                                  float* __restrict__ partial) {
  const int rb    = blockIdx.x >> 2;     // 64 row blocks of 128 rows
  const int split = blockIdx.x & 3;      // 4 column splits of 2048 cols
  const int lane  = threadIdx.x & 31;
  const int ln    = lane & 15;           // M for A / N for B / col in C
  const int h     = lane >> 4;           // half-wave selector
  const int i0    = rb * 128 + (threadIdx.x >> 5) * 16;

  // A fragments: rows i0..i0+15, all of K=128, resident in VGPRs.
  // A 16x32 layout (matches B): lane(ln,h) holds K = 32s + 16h + [0..15].
  const _Float16* arow = r + (size_t)(i0 + ln) * DDIM;
  v16h afrag[4];
#pragma unroll
  for (int s = 0; s < 4; ++s) {
    const int kb = 32 * s;
    const v8h lo = *(const v8h*)(arow + kb + 16 * h);
    const v8h hi = *(const v8h*)(arow + kb + 16 * h + 8);
#pragma unroll
    for (int e = 0; e < 8; ++e) { afrag[s][e] = lo[e]; afrag[s][e + 8] = hi[e]; }
  }

  __shared__ __align__(32) _Float16 sb[2][16 * DDIM];   // 2 x 4 KiB B tiles
  const unsigned lds0 = (unsigned)(uintptr_t)&sb[0][0];
  const unsigned lds1 = (unsigned)(uintptr_t)&sb[1][0];
  const unsigned long long gbase =
      (unsigned long long)(uintptr_t)r + (unsigned long long)split * 2048ull * 256ull;

  float rs[8] = {0.f, 0.f, 0.f, 0.f, 0.f, 0.f, 0.f, 0.f};

  const int NT = 128;                    // 16-column tiles per split
  if (threadIdx.x < 32) {                // wave 0 drives the TDM
    tdm_load_tile(gbase, lds0);
    __builtin_amdgcn_s_wait_tensorcnt(0);
  }
  __syncthreads();

  for (int t = 0; t < NT; ++t) {
    if (t + 1 < NT && threadIdx.x < 32)
      tdm_load_tile(gbase + (unsigned long long)(t + 1) * 4096ull,
                    ((t + 1) & 1) ? lds1 : lds0);

    // B fragment (32x16 layout): lane holds col ln, K = 32s + 16h + [0..15]
    const _Float16* bbase = &sb[t & 1][0] + ln * DDIM + h * 16;
    v8f c = {0.f, 0.f, 0.f, 0.f, 0.f, 0.f, 0.f, 0.f};
#pragma unroll
    for (int s = 0; s < 4; ++s) {
      const v16h b = *(const v16h*)(bbase + 32 * s);
      c = __builtin_amdgcn_wmma_f32_16x16x32_f16(false, afrag[s], false, b,
                                                 (short)0, c, false, false);
    }
#pragma unroll
    for (int v = 0; v < 8; ++v) rs[v] += __expf(2.0f * c[v] - 2.0f);

    if (threadIdx.x < 32) __builtin_amdgcn_s_wait_tensorcnt(0);
    __syncthreads();                     // tile consumed + next tile landed
  }

  // C layout: VGPR v of half h is row (i0 + 8h + v); reduce over the 16 lanes.
#pragma unroll
  for (int m = 1; m <= 8; m <<= 1)
#pragma unroll
    for (int v = 0; v < 8; ++v) rs[v] += __shfl_xor(rs[v], m, 32);
  if (ln == 0) {
    float* dst = partial + (size_t)split * NROWS + i0 + h * 8;
#pragma unroll
    for (int v = 0; v < 8; ++v) dst[v] = rs[v];
  }
}

// ---------------------------------------------------------------------------
// Kernel D: combine partials, apply fixed-shift logsumexp correction, mean.
// ---------------------------------------------------------------------------
__global__ __launch_bounds__(256) void ntx_finish(const float* __restrict__ partial,
                                                  const float* __restrict__ pos,
                                                  const float* __restrict__ dk,
                                                  float* __restrict__ out) {
  __shared__ float red[256];
  float acc = 0.f;
  for (int k = threadIdx.x; k < NROWS; k += 256) {
    const float s = partial[k] + partial[NROWS + k] +
                    partial[2 * NROWS + k] + partial[3 * NROWS + k];
    const float p = pos[k];
    const float d = dk[k];
    // denom = sum_{j != k} exp(2 sim - 2) + exp(2 pos - 2)  (pos counted twice)
    const float denom = s - __expf(2.f * d - 2.f) + __expf(2.f * p - 2.f);
    acc += 2.f - 2.f * p + logf(denom);
  }
  red[threadIdx.x] = acc;
  __syncthreads();
  for (int off = 128; off > 0; off >>= 1) {
    if (threadIdx.x < off) red[threadIdx.x] += red[threadIdx.x + off];
    __syncthreads();
  }
  if (threadIdx.x == 0) out[0] = red[0] * (1.0f / (float)NROWS);
}

extern "C" void kernel_launch(void* const* d_in, const int* in_sizes, int n_in,
                              void* d_out, int out_size, void* d_ws, size_t ws_size,
                              hipStream_t stream) {
  const float* zi = (const float*)d_in[0];
  const float* zj = (const float*)d_in[1];
  char* ws = (char*)d_ws;
  _Float16* r  = (_Float16*)ws;                              // 2 MiB
  float* pos   = (float*)(ws + (2u << 20));                  // 32 KiB
  float* dk    = (float*)(ws + (2u << 20) + (32u << 10));    // 32 KiB
  float* part  = (float*)(ws + (2u << 20) + (64u << 10));    // 128 KiB

  ntx_normalize<<<NROWS / 8, 256, 0, stream>>>(zi, zj, r);
  ntx_posdiag <<<NROWS / 8, 256, 0, stream>>>(r, pos, dk);
  ntx_simsum  <<<64 * 4,    256, 0, stream>>>(r, part);
  ntx_finish  <<<1,         256, 0, stream>>>(part, pos, dk, (float*)d_out);
}
